// MambaSpatioTemporalModel_72524817760635
// MI455X (gfx1250) — matrix operand order repaired and checked by reading, not compiled, not run on hardware
//
#include <hip/hip_runtime.h>
#include <hip/hip_bf16.h>

// ---------------------------------------------------------------------------
// MI455X (gfx1250) Mamba2 spatio-temporal block, round 2.
// All GEMM operands pre-split into bf16 hi/lo pairs so the WMMA inner loop
// contains no conversions: just b128 loads + 12x v_wmma_f32_16x16x32_bf16
// per k-step per wave (4 M-tiles x bf16x3). fp32 accumulate throughout.
// ---------------------------------------------------------------------------

typedef __attribute__((ext_vector_type(16))) __bf16 v16bf;
typedef __attribute__((ext_vector_type(8)))  __bf16 v8bf;
typedef __attribute__((ext_vector_type(8)))  float  v8f;

__device__ __forceinline__ int rowmap(int m, int mode) {
    // mode 0: identity. mode 1: m = (t, hw): t = m>>8, hw = m&255 -> hw*16 + t
    return mode ? ((m & 255) * 16 + (m >> 8)) : m;
}

__device__ __forceinline__ void bsplit(float f, __bf16 &hi, __bf16 &lo) {
    hi = (__bf16)f;                 // RNE round to bf16
    lo = (__bf16)(f - (float)hi);   // residual (|lo| <= 2^-8 |f|)
}

__device__ __forceinline__ v16bf cat16(v8bf a, v8bf b) {
    v16bf r;
    #pragma unroll
    for (int e = 0; e < 8; ++e) { r[e] = a[e]; r[8 + e] = b[e]; }
    return r;
}

// ---------------------------------------------------------------------------
// Elementwise fp32 -> bf16 hi/lo split (for weight matrices).
// ---------------------------------------------------------------------------
__global__ __launch_bounds__(256)
void split_f32_kernel(const float* __restrict__ in, __bf16* __restrict__ hi,
                      __bf16* __restrict__ lo, int len)
{
    for (int i = blockIdx.x * 256 + threadIdx.x; i < len; i += gridDim.x * 256) {
        __bf16 h, l; bsplit(in[i], h, l);
        hi[i] = h; lo[i] = l;
    }
}

// ---------------------------------------------------------------------------
// hidden_states transpose + split: hs[c][m] (c-major) -> A[m][c] bf16 hi/lo,
// m = (t,hw) flat index, 4096 rows x 512 cols. LDS-tiled for coalescing.
// ---------------------------------------------------------------------------
__global__ __launch_bounds__(256)
void split_hs_kernel(const float* __restrict__ hs,
                     __bf16* __restrict__ ahi, __bf16* __restrict__ alo)
{
    __shared__ float tile[32][33];
    const int tid = threadIdx.x;
    const int l   = tid & 31;
    const int bm  = blockIdx.x * 32, bc = blockIdx.y * 32;
    #pragma unroll
    for (int cc = tid >> 5; cc < 32; cc += 8)
        tile[cc][l] = hs[(size_t)(bc + cc) * 4096 + bm + l];   // coalesced over m
    __syncthreads();
    #pragma unroll
    for (int mm = tid >> 5; mm < 32; mm += 8) {
        __bf16 h, lo_; bsplit(tile[l][mm], h, lo_);
        size_t o = (size_t)(bm + mm) * 512 + bc + l;           // coalesced over c
        ahi[o] = h; alo[o] = lo_;
    }
}

// ---------------------------------------------------------------------------
// WMMA GEMM on pre-split bf16 operands:
//   C[m,n] = sum_k A[m,k]*W[n,k] (+bias[n]),  A = Ahi+Alo, W = Whi+Wlo
// A row-major [M][K] (row index via rowmap(a_map)), W row-major [N][K].
// Output: fp32 strided (Cf) and/or bf16 hi/lo pair (Chi/Clo, ldc_bf).
// Block 256 thr (8 waves); tile 64(M) x 128(N); K-step 32; 12 WMMA / step.
// ---------------------------------------------------------------------------
__global__ __launch_bounds__(256)
void wmma_gemm_bf(const __bf16* __restrict__ Ahi, const __bf16* __restrict__ Alo,
                  int a_map,
                  const __bf16* __restrict__ Whi, const __bf16* __restrict__ Wlo,
                  const float* __restrict__ bias,
                  float* __restrict__ Cf, int c_rs, int c_cs,
                  __bf16* __restrict__ Chi, __bf16* __restrict__ Clo, int ldc_bf,
                  int c_map, int M, int N, int K)
{
    __shared__ __bf16 Ah[64 * 32];
    __shared__ __bf16 Al[64 * 32];

    const int tid    = threadIdx.x;
    const int wave   = tid >> 5;
    const int lane   = tid & 31;
    const int laneHi = lane >> 4;
    const int lmod   = lane & 15;
    const int m0     = blockIdx.x * 64;
    const int n0     = (blockIdx.y * 8 + wave) * 16;
    const bool active = (n0 < N);

    v8f acc[4];
    #pragma unroll
    for (int mt = 0; mt < 4; ++mt) acc[mt] = (v8f){0.f,0.f,0.f,0.f,0.f,0.f,0.f,0.f};

    const int si = tid >> 2;           // staging row 0..63
    const int sj = (tid & 3) * 8;      // staging col chunk (8 bf16 = 16B)
    const long srow = (long)rowmap(m0 + si, a_map) * K;

    for (int k0 = 0; k0 < K; k0 += 32) {
        __syncthreads();
        // cooperative A stage: one b128 global load + one b128 ds store per array
        *(v8bf*)(Ah + si * 32 + sj) = *(const v8bf*)(Ahi + srow + k0 + sj);
        *(v8bf*)(Al + si * 32 + sj) = *(const v8bf*)(Alo + srow + k0 + sj);
        if (k0 + 32 < K) {
            __builtin_prefetch(Ahi + srow + k0 + 32 + sj, 0, 1);   // global_prefetch_b8
            __builtin_prefetch(Alo + srow + k0 + 32 + sj, 0, 1);
        }
        __syncthreads();

        if (active) {
            // B operand: lane = N, K runs {8*laneHi..+7} and {16+8*laneHi..+7}
            const __bf16* wh = Whi + (long)(n0 + lmod) * K + k0 + 8 * laneHi;
            const __bf16* wl = Wlo + (long)(n0 + lmod) * K + k0 + 8 * laneHi;
            v16bf bhi = cat16(*(const v8bf*)wh, *(const v8bf*)(wh + 16));
            v16bf blo = cat16(*(const v8bf*)wl, *(const v8bf*)(wl + 16));

            #pragma unroll
            for (int mt = 0; mt < 4; ++mt) {
                // A operand: lane = M, K runs {8*laneHi..+7} and {16+8*laneHi..+7}
                const int row = (mt * 16 + lmod) * 32;
                v16bf ahi = cat16(*(const v8bf*)(Ah + row + 8 * laneHi),
                                  *(const v8bf*)(Ah + row + 16 + 8 * laneHi));
                v16bf alo = cat16(*(const v8bf*)(Al + row + 8 * laneHi),
                                  *(const v8bf*)(Al + row + 16 + 8 * laneHi));
                acc[mt] = __builtin_amdgcn_wmma_f32_16x16x32_bf16(false, ahi, false, bhi, (short)0, acc[mt], false, false);
                acc[mt] = __builtin_amdgcn_wmma_f32_16x16x32_bf16(false, alo, false, bhi, (short)0, acc[mt], false, false);
                acc[mt] = __builtin_amdgcn_wmma_f32_16x16x32_bf16(false, ahi, false, blo, (short)0, acc[mt], false, false);
            }
        }
    }

    if (active) {
        const int   n  = n0 + lmod;
        const float bv = bias ? bias[n] : 0.f;
        #pragma unroll
        for (int mt = 0; mt < 4; ++mt) {
            #pragma unroll
            for (int v = 0; v < 8; ++v) {
                // C/D f32 layout: VGPR v -> M = v + 8*laneHi, N = lane%16
                int   m   = m0 + mt * 16 + v + 8 * laneHi;
                int   r   = rowmap(m, c_map);
                float val = acc[mt][v] + bv;
                if (Cf)  Cf[(long)r * c_rs + (long)n * c_cs] = val;
                if (Chi) {
                    __bf16 h, l; bsplit(val, h, l);
                    Chi[(long)r * ldc_bf + n] = h;
                    Clo[(long)r * ldc_bf + n] = l;
                }
            }
        }
    }
}

// ---------------------------------------------------------------------------
// Fused conv1d + SiLU + selective scan + gating + RMSnorm.
// One block per sequence n (256 blocks), 512 threads; 64-float state in regs.
// Emits normalized y directly as bf16 hi/lo for the out_proj WMMA GEMM.
// ---------------------------------------------------------------------------
__global__ __launch_bounds__(512)
void mamba_core(const float* __restrict__ zx,      // [256*16, 2192] fp32
                const float* __restrict__ conv_w,  // [1152, 4]
                const float* __restrict__ conv_b,  // [1152]
                const float* __restrict__ dt_bias, // [16]
                const float* __restrict__ A_log,   // [16]
                const float* __restrict__ Dp,      // [16]
                const float* __restrict__ norm_w,  // [1024]
                __bf16* __restrict__ ynh,          // [256*16, 1024] hi
                __bf16* __restrict__ ynl)          // [256*16, 1024] lo
{
    __shared__ float bc[16][128];   // conv+silu of B (0..63) and C (64..127) per t
    __shared__ float dts[16][16];   // softplus dt per (t, head)
    __shared__ float red[16];

    const int n   = blockIdx.x;
    const int tid = threadIdx.x;
    const long base = (long)n * 16 * 2192;

    // ---- Phase 0a: conv+silu for B/C channels (xBC channels 1024..1151) ----
    for (int idx = tid; idx < 2048; idx += 512) {
        int t = idx >> 7, cc = idx & 127;
        int c = 1024 + cc;
        float acc = conv_b[c];
        #pragma unroll
        for (int k = 0; k < 4; ++k) {
            int tt = t + k - 3;
            if (tt >= 0) acc += zx[base + (long)tt * 2192 + 1024 + c] * conv_w[c * 4 + k];
        }
        bc[t][cc] = acc / (1.f + expf(-acc));      // silu
    }
    // ---- Phase 0b: dt = softplus(dt_raw + dt_bias) ----
    if (tid < 256) {
        int t = tid >> 4, h = tid & 15;
        float v = zx[base + (long)t * 2192 + 2176 + h] + dt_bias[h];
        dts[t][h] = (v > 20.f) ? v : log1pf(expf(v));
    }
    __syncthreads();

    // ---- Phase 1: per-channel conv + scan, 2 jobs per thread ----
    float yv[2][16];
    #pragma unroll
    for (int j = 0; j < 2; ++j) {
        const int d    = tid + j * 512;
        const int head = d >> 6;
        const float Aneg = -expf(A_log[head]);
        const float Dh   = Dp[head];

        float raw[16];
        #pragma unroll
        for (int t = 0; t < 16; ++t) raw[t] = zx[base + (long)t * 2192 + 1024 + d];

        const float w0 = conv_w[d * 4 + 0], w1 = conv_w[d * 4 + 1];
        const float w2 = conv_w[d * 4 + 2], w3 = conv_w[d * 4 + 3];
        const float cb = conv_b[d];

        float xs[16];
        #pragma unroll
        for (int t = 0; t < 16; ++t) {
            float acc = cb + raw[t] * w3;
            if (t >= 1) acc += raw[t - 1] * w2;
            if (t >= 2) acc += raw[t - 2] * w1;
            if (t >= 3) acc += raw[t - 3] * w0;
            xs[t] = acc / (1.f + expf(-acc));      // silu
        }

        float st[64];
        #pragma unroll
        for (int s = 0; s < 64; ++s) st[s] = 0.f;

        #pragma unroll
        for (int t = 0; t < 16; ++t) {
            float dt    = dts[t][head];
            float dA    = expf(dt * Aneg);
            float coeff = dt * xs[t];
            float y = 0.f;
            #pragma unroll
            for (int s = 0; s < 64; ++s) {
                st[s] = st[s] * dA + coeff * bc[t][s];   // B_t
                y    += st[s] * bc[t][64 + s];           // C_t
            }
            yv[j][t] = y + Dh * xs[t];
        }
    }

    // ---- Phase 2: gate with silu(z), RMS norm over 1024, write bf16 pair ----
    const int lane = tid & 31, wid = tid >> 5;     // 16 waves (wave32)
    #pragma unroll
    for (int t = 0; t < 16; ++t) {
        float vals[2], ss = 0.f;
        #pragma unroll
        for (int j = 0; j < 2; ++j) {
            int d = tid + j * 512;
            float z   = zx[base + (long)t * 2192 + d];
            float val = yv[j][t] * (z / (1.f + expf(-z)));
            vals[j] = val;
            ss += val * val;
        }
        #pragma unroll
        for (int off = 16; off >= 1; off >>= 1) ss += __shfl_xor(ss, off, 32);
        if (lane == 0) red[wid] = ss;
        __syncthreads();
        if (tid < 16) {
            float s2 = red[tid];
            #pragma unroll
            for (int off = 8; off >= 1; off >>= 1) s2 += __shfl_xor(s2, off, 16);
            if (tid == 0) red[0] = rsqrtf(s2 * (1.f / 1024.f) + 1e-5f);
        }
        __syncthreads();
        float scale = red[0];
        #pragma unroll
        for (int j = 0; j < 2; ++j) {
            int d = tid + j * 512;
            __bf16 h, l; bsplit(vals[j] * scale * norm_w[d], h, l);
            long o = (long)(n * 16 + t) * 1024 + d;
            ynh[o] = h; ynl[o] = l;
        }
        __syncthreads();
    }
}

// ---------------------------------------------------------------------------
// Launcher. Input order: hidden_states, proj_in_w, proj_in_b, proj_out_w,
// proj_out_b, in_proj_w, conv_w, conv_b, dt_bias, A_log, D, norm_w, out_proj_w.
// ---------------------------------------------------------------------------
extern "C" void kernel_launch(void* const* d_in, const int* in_sizes, int n_in,
                              void* d_out, int out_size, void* d_ws, size_t ws_size,
                              hipStream_t stream) {
    const float* hs         = (const float*)d_in[0];
    const float* proj_in_w  = (const float*)d_in[1];
    const float* proj_in_b  = (const float*)d_in[2];
    const float* proj_out_w = (const float*)d_in[3];
    const float* proj_out_b = (const float*)d_in[4];
    const float* in_proj_w  = (const float*)d_in[5];
    const float* conv_w     = (const float*)d_in[6];
    const float* conv_b     = (const float*)d_in[7];
    const float* dt_bias    = (const float*)d_in[8];
    const float* A_log      = (const float*)d_in[9];
    const float* Dp         = (const float*)d_in[10];
    const float* norm_w     = (const float*)d_in[11];
    const float* out_proj_w = (const float*)d_in[12];

    // workspace carve-up (256B aligned chunks)
    char* p = (char*)d_ws;
    auto carve = [&](size_t bytes) { void* r = (void*)p; p += (bytes + 255) & ~(size_t)255; return r; };

    float*  zxb  = (float*) carve((size_t)4096 * 2192 * 4);   // fp32 zxbcdt, rows (hw,t)
    __bf16* hshi = (__bf16*)carve((size_t)4096 * 512 * 2);    // hs transposed, rows (t,hw)
    __bf16* hslo = (__bf16*)carve((size_t)4096 * 512 * 2);
    __bf16* xhi  = (__bf16*)carve((size_t)4096 * 512 * 2);    // x, rows (hw,t)
    __bf16* xlo  = (__bf16*)carve((size_t)4096 * 512 * 2);
    __bf16* ynhi = (__bf16*)carve((size_t)4096 * 1024 * 2);   // normalized y, rows (hw,t)
    __bf16* ynlo = (__bf16*)carve((size_t)4096 * 1024 * 2);
    __bf16* yphi = (__bf16*)carve((size_t)4096 * 512 * 2);    // out_proj result, rows (hw,t)
    __bf16* yplo = (__bf16*)carve((size_t)4096 * 512 * 2);
    __bf16* w1hi = (__bf16*)carve((size_t)512 * 512 * 2);     // proj_in_w
    __bf16* w1lo = (__bf16*)carve((size_t)512 * 512 * 2);
    __bf16* w2hi = (__bf16*)carve((size_t)2192 * 512 * 2);    // in_proj_w
    __bf16* w2lo = (__bf16*)carve((size_t)2192 * 512 * 2);
    __bf16* w4hi = (__bf16*)carve((size_t)512 * 1024 * 2);    // out_proj_w
    __bf16* w4lo = (__bf16*)carve((size_t)512 * 1024 * 2);
    __bf16* w5hi = (__bf16*)carve((size_t)512 * 512 * 2);     // proj_out_w
    __bf16* w5lo = (__bf16*)carve((size_t)512 * 512 * 2);

    // 0) split weights and transpose+split hidden_states
    split_f32_kernel<<<256, 256, 0, stream>>>(proj_in_w,  w1hi, w1lo, 512 * 512);
    split_f32_kernel<<<512, 256, 0, stream>>>(in_proj_w,  w2hi, w2lo, 2192 * 512);
    split_f32_kernel<<<256, 256, 0, stream>>>(out_proj_w, w4hi, w4lo, 512 * 1024);
    split_f32_kernel<<<256, 256, 0, stream>>>(proj_out_w, w5hi, w5lo, 512 * 512);
    split_hs_kernel<<<dim3(128, 16), 256, 0, stream>>>(hs, hshi, hslo);

    // 1) proj_in: x[(hw,t), d] = sum_c hsT[(t,hw), c] * W1[d,c] + b -> bf16 pair
    wmma_gemm_bf<<<dim3(64, 4), 256, 0, stream>>>(
        hshi, hslo, 0, w1hi, w1lo, proj_in_b,
        nullptr, 0, 0, xhi, xlo, 512, /*c_map=*/1, 4096, 512, 512);

    // 2) in_proj: zxbcdt[(hw,t), e] = sum_d x * W2[e,d] -> fp32 (scan input)
    wmma_gemm_bf<<<dim3(64, 18), 256, 0, stream>>>(
        xhi, xlo, 0, w2hi, w2lo, nullptr,
        zxb, 2192, 1, nullptr, nullptr, 0, 0, 4096, 2192, 512);

    // 3) conv + scan + gate + RMSnorm -> bf16 pair
    mamba_core<<<256, 512, 0, stream>>>(zxb, conv_w, conv_b, dt_bias, A_log, Dp,
                                        norm_w, ynhi, ynlo);

    // 4) out_proj: yp[(hw,t), d] = sum_k yn * W4[d,k] -> bf16 pair
    wmma_gemm_bf<<<dim3(64, 4), 256, 0, stream>>>(
        ynhi, ynlo, 0, w4hi, w4lo, nullptr,
        nullptr, 0, 0, yphi, yplo, 512, 0, 4096, 512, 1024);

    // 5) proj_out: out[c, (t,hw)] = sum_d yp[(hw,t), d] * W5[c,d] + b -> fp32
    //    A rows permuted (m=(t,hw) -> hw*16+t); C strided (c_rs=1, c_cs=4096).
    wmma_gemm_bf<<<dim3(64, 4), 256, 0, stream>>>(
        yphi, yplo, 1, w5hi, w5lo, proj_out_b,
        (float*)d_out, 1, 4096, nullptr, nullptr, 0, 0, 4096, 512, 512);
}
